// SNNLanguageModel_10763188043848
// MI455X (gfx1250) — matrix operand order repaired and checked by reading, not compiled, not run on hardware
//
#include <hip/hip_runtime.h>

// ---------------------------------------------------------------------------
// SNN language model for MI455X (gfx1250, wave32, WMMA).
//   T=128, B=32, H=512, V=50257.  Dominant op: [4096,512] x [V,512]^T GEMM.
//   bf16 WMMA (v_wmma_f32_16x16x32_bf16), f32 accumulate.
//   - 51MB bf16 weights stay L2-resident; output stores are non-temporal so
//     the 823MB logits stream does not evict them.
//   - Each wave computes 32x64; the 8 waves of a block stack in M over one
//     64-col B strip, so B fragments hit the WGP$ (L2 B traffic ~0.8GB total).
// ---------------------------------------------------------------------------

#define TT 128
#define BB 32
#define HH 512
#define VV 50257
#define VPAD 50304   // 786 * 64, multiple of the 64-wide N macro-tile

typedef __attribute__((ext_vector_type(16))) __bf16 v16bf;
typedef __attribute__((ext_vector_type(8)))  float  v8f;
typedef __attribute__((ext_vector_type(4)))  unsigned int u32x4;

union Frag16 {           // 32 bytes = 16 bf16 = 8 VGPRs (one WMMA A/B operand)
    v16bf v;
    u32x4 q[2];
};

// ---------------------------------------------------------------------------
// fp32 -> bf16 conversion with zero padding (pads fco_w rows V..VPAD)
// ---------------------------------------------------------------------------
__global__ void cvt_f32_to_bf16(const float* __restrict__ in,
                                __bf16* __restrict__ out,
                                int nvalid, int ntotal) {
    int i = blockIdx.x * blockDim.x + threadIdx.x;
    if (i < ntotal) {
        float v = (i < nvalid) ? in[i] : 0.0f;
        out[i] = (__bf16)v;
    }
}

// ---------------------------------------------------------------------------
// Token embed + positional embed + leaky scan 1  ->  bf16 activations [T*B, H]
// one thread per (b,h) channel; scan over T in a register.
// ---------------------------------------------------------------------------
__global__ void embed_scan1(const int* __restrict__ x,        // [T,B]
                            const float* __restrict__ emb,    // [V,H]
                            const float* __restrict__ pos,    // [T,H]
                            const float* __restrict__ beta,   // [H]
                            __bf16* __restrict__ out) {       // [T*B, H] bf16
    int idx = blockIdx.x * blockDim.x + threadIdx.x;          // [0, B*H)
    int b = idx / HH;
    int h = idx - b * HH;
    float bt  = beta[h];
    float mem = 0.0f;
    for (int t = 0; t < TT; ++t) {
        int tok  = x[t * BB + b];
        float v  = emb[(size_t)tok * HH + h] + pos[t * HH + h];
        mem = bt * mem + v;
        out[(size_t)(t * BB + b) * HH + h] = (__bf16)mem;
    }
}

// ---------------------------------------------------------------------------
// Leaky scan over a fp32 [T,B,H] tensor -> bf16 activations for next GEMM
// ---------------------------------------------------------------------------
__global__ void scan_to_bf16(const float* __restrict__ in,    // [T*B*H] f32
                             const float* __restrict__ beta,  // [H]
                             __bf16* __restrict__ out) {      // [T*B*H] bf16
    int idx = blockIdx.x * blockDim.x + threadIdx.x;          // [0, B*H)
    int h = idx & (HH - 1);
    float bt  = beta[h];
    float mem = 0.0f;
    for (int t = 0; t < TT; ++t) {
        float v = in[(size_t)t * (BB * HH) + idx];
        mem = bt * mem + v;
        out[(size_t)t * (BB * HH) + idx] = (__bf16)mem;
    }
}

// ---------------------------------------------------------------------------
// bf16 WMMA GEMM:  C[M,Nvalid] = A[M,K] * W[Npad,K]^T (+bias) (optional ReLU)
// Wave tile: 32x64 (2 A fragments x 4 B fragments -> 8 WMMA accumulators).
// Block = 8 waves stacked in M over the same 64-wide B strip (B hits WGP$).
// K stepped by 32 (v_wmma_f32_16x16x32_bf16). Non-temporal C stores.
// ---------------------------------------------------------------------------
__global__ void __launch_bounds__(256)
gemm_bf16_wmma(const __bf16* __restrict__ A,    // [M,K] row-major
               const __bf16* __restrict__ W,    // [Npad,K] row-major
               const float* __restrict__ bias,  // [Nvalid]
               float* __restrict__ C,           // [M,Nvalid]
               int M, int K, int Nvalid,
               int numWGm,                      // ceil(M / 256)
               int relu) {
    const int lane = threadIdx.x & 31;
    const int wv   = threadIdx.x >> 5;                  // wave in block: 0..7
    const int wgn  = blockIdx.x / numWGm;               // 64-col strip index
    const int wgm  = blockIdx.x % numWGm;               // 256-row super-tile
    const int m0   = (wgm << 8) + (wv << 5);            // this wave: 32 rows
    const int n0   = wgn << 6;                          // this block: 64 cols
    if (m0 >= M) return;                                // wave-uniform exit

    const int hi = lane >> 4;                           // half-wave select
    const int lr = lane & 15;

    // A fragments (ISA 16-bit A layout):
    //   lane<16 : row=lr, K = kk+0..7  and kk+16..23
    //   lane>=16: row=lr, K = kk+8..15 and kk+24..31
    const __bf16* arow0 = A + (size_t)(m0 + lr) * K + (hi ? 8 : 0);
    const __bf16* arow1 = arow0 + (size_t)16 * K;

    // B fragments (column n of B = row n of W):
    //   lane<16 : n = n0+t*16+lr, K = kk+0..7  and kk+16..23
    //   lane>=16:                 K = kk+8..15 and kk+24..31
    const __bf16* brow = W + (size_t)(n0 + lr) * K + (hi ? 8 : 0);

    v8f acc[8] = {};

    for (int kk = 0; kk < K; kk += 32) {
        Frag16 a0, a1;
        a0.q[0] = *(const u32x4*)(arow0 + kk);
        a0.q[1] = *(const u32x4*)(arow0 + kk + 16);
        a1.q[0] = *(const u32x4*)(arow1 + kk);
        a1.q[1] = *(const u32x4*)(arow1 + kk + 16);

#pragma unroll
        for (int t = 0; t < 4; ++t) {
            Frag16 b;
            const __bf16* bp = brow + (size_t)(t * 16) * K + kk;
            b.q[0] = *(const u32x4*)(bp);
            b.q[1] = *(const u32x4*)(bp + 16);
            acc[t]     = __builtin_amdgcn_wmma_f32_16x16x32_bf16(
                             false, a0.v, false, b.v, (short)0, acc[t], false, false);
            acc[4 + t] = __builtin_amdgcn_wmma_f32_16x16x32_bf16(
                             false, a1.v, false, b.v, (short)0, acc[4 + t], false, false);
        }
    }

    // Epilogue: C/D layout -> VGPR r holds (row m0+r(+8 for hi), col lane&15).
    // Non-temporal: don't let the output stream evict L2-resident weights.
    const int rb0 = m0 + (hi ? 8 : 0);
    const int rb1 = rb0 + 16;
#pragma unroll
    for (int t = 0; t < 4; ++t) {
        int col = n0 + (t << 4) + lr;
        if (col < Nvalid) {
            float bv = bias[col];
#pragma unroll
            for (int r = 0; r < 8; ++r) {
                float v0 = acc[t][r] + bv;
                float v1 = acc[4 + t][r] + bv;
                if (relu) { v0 = fmaxf(v0, 0.0f); v1 = fmaxf(v1, 0.0f); }
                __builtin_nontemporal_store(v0, &C[(size_t)(rb0 + r) * Nvalid + col]);
                __builtin_nontemporal_store(v1, &C[(size_t)(rb1 + r) * Nvalid + col]);
            }
        }
    }
}

// ---------------------------------------------------------------------------
// Launch
// ---------------------------------------------------------------------------
extern "C" void kernel_launch(void* const* d_in, const int* in_sizes, int n_in,
                              void* d_out, int out_size, void* d_ws, size_t ws_size,
                              hipStream_t stream) {
    (void)in_sizes; (void)n_in; (void)out_size; (void)ws_size;

    const int*   x      = (const int*)  d_in[0];
    const float* emb    = (const float*)d_in[1];
    const float* pos    = (const float*)d_in[2];
    const float* beta1  = (const float*)d_in[3];
    const float* beta2  = (const float*)d_in[4];
    const float* beta3  = (const float*)d_in[5];
    const float* fc2_w  = (const float*)d_in[6];
    const float* fc2_b  = (const float*)d_in[7];
    const float* fc3_w  = (const float*)d_in[8];
    const float* fc3_b  = (const float*)d_in[9];
    const float* fco_w  = (const float*)d_in[10];
    const float* fco_b  = (const float*)d_in[11];
    float* out = (float*)d_out;

    // workspace carve-up (256B aligned)
    char* ws = (char*)d_ws;
    size_t off = 0;
    auto carve = [&](size_t bytes) {
        char* p = ws + off;
        off += (bytes + 255) & ~(size_t)255;
        return p;
    };
    const size_t MB_act = (size_t)TT * BB * HH;        // 2,097,152 elements
    __bf16* A1 = (__bf16*)carve(MB_act * 2);           // scan1 out (bf16)
    float*  h2 = (float*) carve(MB_act * 4);           // fc2 out (f32)
    __bf16* A2 = (__bf16*)carve(MB_act * 2);           // scan2 out
    float*  h3 = (float*) carve(MB_act * 4);           // fc3 out
    __bf16* A3 = (__bf16*)carve(MB_act * 2);           // scan3 out
    __bf16* w2 = (__bf16*)carve((size_t)HH * HH * 2);  // fc2_w bf16
    __bf16* w3 = (__bf16*)carve((size_t)HH * HH * 2);  // fc3_w bf16
    __bf16* wo = (__bf16*)carve((size_t)VPAD * HH * 2);// fco_w bf16 (padded)

    const int M      = TT * BB;            // 4096
    const int numWGm = M / 256;            // 16 (M divisible by 256)

    // 1) weight conversions
    {
        int n = HH * HH;
        cvt_f32_to_bf16<<<(n + 255) / 256, 256, 0, stream>>>(fc2_w, w2, n, n);
        cvt_f32_to_bf16<<<(n + 255) / 256, 256, 0, stream>>>(fc3_w, w3, n, n);
        int ntot = VPAD * HH, nval = VV * HH;
        cvt_f32_to_bf16<<<(ntot + 255) / 256, 256, 0, stream>>>(fco_w, wo, nval, ntot);
    }

    // 2) embed + scan1
    embed_scan1<<<(BB * HH) / 256, 256, 0, stream>>>(x, emb, pos, beta1, A1);

    // 3) fc2 GEMM (+bias, ReLU)
    {
        int tilesN = HH / 64;                               // 8
        gemm_bf16_wmma<<<numWGm * tilesN, 256, 0, stream>>>(
            A1, w2, fc2_b, h2, M, HH, HH, numWGm, 1);
    }

    // 4) scan2
    scan_to_bf16<<<(BB * HH) / 256, 256, 0, stream>>>(h2, beta2, A2);

    // 5) fc3 GEMM (+bias, ReLU)
    {
        int tilesN = HH / 64;
        gemm_bf16_wmma<<<numWGm * tilesN, 256, 0, stream>>>(
            A2, w3, fc3_b, h3, M, HH, HH, numWGm, 1);
    }

    // 6) scan3
    scan_to_bf16<<<(BB * HH) / 256, 256, 0, stream>>>(h3, beta3, A3);

    // 7) output projection GEMM (+bias) -> logits [T,B,V] fp32
    {
        int tilesN = VPAD / 64;                             // 786
        gemm_bf16_wmma<<<numWGm * tilesN, 256, 0, stream>>>(
            A3, wo, fco_b, out, M, HH, VV, numWGm, 0);
    }
}
